// QuadLoss_6811818131843
// MI455X (gfx1250) — matrix operand order repaired and checked by reading, not compiled, not run on hardware
//
#include <hip/hip_runtime.h>
#include <hip/hip_bf16.h>
#include <math.h>

// ---------------------------------------------------------------------------
// QuadLoss (hard-mining triplet loss) for MI455X / gfx1250.
// dist = sqrt(clip(||xi||^2 + ||xj||^2 - 2 xi.xj, 1e-12, 1e12)) via split-bf16
// WMMA (hi*hi + lo*hi + hi*lo accumulated in f32 with v_wmma_f32_16x16x32_bf16).
// Epilogue transposes each 16x16 WMMA tile through LDS so the 256 MB dist
// write stream is fully 128B-coalesced (the HBM roofline term).
// ---------------------------------------------------------------------------

typedef __bf16 bf16_t;
typedef __attribute__((ext_vector_type(16))) __bf16 v16bf;
typedef __attribute__((ext_vector_type(8)))  __bf16 v8bf;
typedef __attribute__((ext_vector_type(8)))  float  v8f;

#define QL_MARGIN 0.3f
#define QL_BIG    1e12f
#define QL_EPS    1e-12f

// LDS tile: 32 rows x 64 cols, padded row stride 68 floats (write-conflict-free)
#define QL_LDS_STRIDE 68

__device__ __forceinline__ bf16_t ql_f32_to_bf16(float x) {
  // round-to-nearest-even truncation to bf16 (inputs are finite normals)
  unsigned u = __builtin_bit_cast(unsigned, x);
  unsigned r = (u + 0x7FFFu + ((u >> 16) & 1u)) >> 16;
  unsigned short s = (unsigned short)r;
  return __builtin_bit_cast(bf16_t, s);
}
__device__ __forceinline__ float ql_bf16_to_f32(bf16_t h) {
  unsigned short s = __builtin_bit_cast(unsigned short, h);
  unsigned u = ((unsigned)s) << 16;
  return __builtin_bit_cast(float, u);
}

// ---------------------------------------------------------------------------
// Kernel 1: sq[i] = sum_k x[i,k]^2 ; split X into bf16 hi/lo parts.
// grid = n rows, block = d (=128) threads.
// ---------------------------------------------------------------------------
__global__ void ql_prep_kernel(const float* __restrict__ X,
                               bf16_t* __restrict__ Xhi,
                               bf16_t* __restrict__ Xlo,
                               float* __restrict__ sq, int d) {
  const int row = blockIdx.x;
  const int t   = threadIdx.x;
  const float x = X[(size_t)row * d + t];
  const bf16_t h = ql_f32_to_bf16(x);
  const float  hf = ql_bf16_to_f32(h);
  const bf16_t l = ql_f32_to_bf16(x - hf);
  Xhi[(size_t)row * d + t] = h;
  Xlo[(size_t)row * d + t] = l;

  __shared__ float s[128];
  s[t] = x * x;
  __syncthreads();
  for (int off = 64; off > 0; off >>= 1) {
    if (t < off) s[t] += s[t + off];
    __syncthreads();
  }
  if (t == 0) sq[row] = s[0];
}

// ---------------------------------------------------------------------------
// Kernel 2: tiled X*X^T via bf16 split WMMA + dist epilogue.
// Block = 256 threads = 8 waves; each wave owns one 16x16 tile; block covers
// a 32(row) x 64(col) region. grid = (n/64, n/32).
// Output goes through LDS so global stores are fully coalesced b128s.
// ---------------------------------------------------------------------------
__global__ __launch_bounds__(256, 2)
void ql_gemm_dist_kernel(const bf16_t* __restrict__ Xhi,
                         const bf16_t* __restrict__ Xlo,
                         const float*  __restrict__ sq,
                         float* __restrict__ dist, int n, int d) {
  __shared__ float tile[32 * QL_LDS_STRIDE];

  const int lane   = threadIdx.x & 31;
  const int wv     = threadIdx.x >> 5;
  const int laneLo = lane & 15;
  const int laneHi = lane >> 4;          // 0 or 1 (wave32 half)
  const int wr     = wv >> 2;            // 0..1  row-tile within block
  const int wc     = wv & 3;             // 0..3  col-tile within block
  const int itile  = blockIdx.y * 32 + wr * 16;
  const int jtile  = blockIdx.x * 64 + wc * 16;

  const int m    = itile + laneLo;       // A-matrix row held by this lane
  const int ncol = jtile + laneLo;       // B-matrix column held by this lane

  const bf16_t* __restrict__ arow_hi = Xhi + (size_t)m    * d;
  const bf16_t* __restrict__ arow_lo = Xlo + (size_t)m    * d;
  const bf16_t* __restrict__ brow_hi = Xhi + (size_t)ncol * d;
  const bf16_t* __restrict__ brow_lo = Xlo + (size_t)ncol * d;

  v8f acc = {};  // 16x16 f32 accumulator (8 VGPRs per lane, wave32)

  union V16 { v16bf v; v8bf h[2]; };

  #pragma unroll
  for (int k0 = 0; k0 < 128; k0 += 32) {
    // ISA 16-bit A 16x32 layout: lanes 0-15 hold K=0..7 (elems 0..7) and
    // K=16..23 (elems 8..15); lanes 16-31 hold K=8..15 and K=24..31.
    const int ak = k0 + laneHi * 8;
    V16 ahi, alo, bhi, blo;
    ahi.h[0] = *(const v8bf*)(arow_hi + ak);
    ahi.h[1] = *(const v8bf*)(arow_hi + ak + 16);
    alo.h[0] = *(const v8bf*)(arow_lo + ak);
    alo.h[1] = *(const v8bf*)(arow_lo + ak + 16);
    // B 32x16 layout: same K order as A: lanes 0-15 hold K=0..7 and K=16..23,
    // lanes 16-31 hold K=8..15 and K=24..31.
    const int bk = k0 + laneHi * 8;
    bhi.h[0] = *(const v8bf*)(brow_hi + bk);
    bhi.h[1] = *(const v8bf*)(brow_hi + bk + 16);
    blo.h[0] = *(const v8bf*)(brow_lo + bk);
    blo.h[1] = *(const v8bf*)(brow_lo + bk + 16);

    __builtin_prefetch(brow_hi + bk + 32, 0, 3);  // global_prefetch_b8

    // split-precision product: hi*hi + lo*hi + hi*lo (f32 accumulate)
    acc = __builtin_amdgcn_wmma_f32_16x16x32_bf16(false, ahi.v, false, bhi.v,
                                                  (short)0, acc, false, false);
    acc = __builtin_amdgcn_wmma_f32_16x16x32_bf16(false, alo.v, false, bhi.v,
                                                  (short)0, acc, false, false);
    acc = __builtin_amdgcn_wmma_f32_16x16x32_bf16(false, ahi.v, false, blo.v,
                                                  (short)0, acc, false, false);
  }

  // Epilogue part 1: finish dist per lane, scatter into LDS.
  // C/D layout: VGPR v -> M = laneHi*8 + v, N = laneLo.
  const float sqn = sq[ncol];
  const int lr0 = wr * 16 + laneHi * 8;        // row within block's 32x64 tile
  const int lc  = wc * 16 + laneLo;            // col within block's 32x64 tile
  #pragma unroll
  for (int v = 0; v < 8; ++v) {
    const int mr = itile + laneHi * 8 + v;
    float dd = sq[mr] + sqn - 2.0f * acc[v];
    dd = fminf(fmaxf(dd, QL_EPS), QL_BIG);
    tile[(lr0 + v) * QL_LDS_STRIDE + lc] = sqrtf(dd);
  }
  __syncthreads();

  // Epilogue part 2: coalesced writeback. Thread t handles 8 consecutive
  // floats of row (t>>3): consecutive threads -> consecutive 32B chunks.
  const int t  = threadIdx.x;
  const int r  = t >> 3;                       // 0..31
  const int c  = (t & 7) * 8;                  // 0..56
  const int row0 = blockIdx.y * 32;
  const int col0 = blockIdx.x * 64;
  const float4 lo4 = *(const float4*)&tile[r * QL_LDS_STRIDE + c];
  const float4 hi4 = *(const float4*)&tile[r * QL_LDS_STRIDE + c + 4];
  float4* gp = (float4*)(dist + (size_t)(row0 + r) * n + col0 + c);
  gp[0] = lo4;
  gp[1] = hi4;
}

// ---------------------------------------------------------------------------
// Kernel 3: per-row hardest positive / hardest negative (+argmin, first
// occurrence) and rank of the argmin among negatives. grid = n, block = 256.
// ---------------------------------------------------------------------------
__global__ void ql_row_reduce_kernel(const float* __restrict__ dist,
                                     const int* __restrict__ tgt,
                                     float* __restrict__ ap,
                                     float* __restrict__ an,
                                     int* __restrict__ n1, int n) {
  const int i  = blockIdx.x;
  const int t  = threadIdx.x;
  const int ti = tgt[i];
  const float* __restrict__ row = dist + (size_t)i * n;

  float bmax = -QL_BIG;
  float bmin =  QL_BIG;
  int   bidx = 0x7fffffff;
  for (int j = t; j < n; j += 256) {
    const float dv = row[j];
    if (tgt[j] == ti) {
      bmax = fmaxf(bmax, dv);
    } else if (dv < bmin) {           // strict '<' keeps first occurrence
      bmin = dv; bidx = j;
    }
  }

  __shared__ float smax[256];
  __shared__ float smin[256];
  __shared__ int   sidx[256];
  smax[t] = bmax; smin[t] = bmin; sidx[t] = bidx;
  __syncthreads();
  for (int off = 128; off > 0; off >>= 1) {
    if (t < off) {
      smax[t] = fmaxf(smax[t], smax[t + off]);
      const float v2 = smin[t + off];
      const int   i2 = sidx[t + off];
      if (v2 < smin[t] || (v2 == smin[t] && i2 < sidx[t])) {
        smin[t] = v2; sidx[t] = i2;
      }
    }
    __syncthreads();
  }
  const int col = sidx[0];

  // n1_ind = (# of j <= col with tgt[j] != ti) - 1   (rank among negatives)
  int cnt = 0;
  for (int j = t; j <= col; j += 256) cnt += (tgt[j] != ti) ? 1 : 0;
  __shared__ int scnt[256];
  scnt[t] = cnt;
  __syncthreads();
  for (int off = 128; off > 0; off >>= 1) {
    if (t < off) scnt[t] += scnt[t + off];
    __syncthreads();
  }
  if (t == 0) { ap[i] = smax[0]; an[i] = smin[0]; n1[i] = scnt[0] - 1; }
}

// ---------------------------------------------------------------------------
// Kernel 4: dist_n12[i] = min over negatives-of-row-r of dist[r, :], r=n1[i].
// ---------------------------------------------------------------------------
__global__ void ql_n12_kernel(const float* __restrict__ dist,
                              const int* __restrict__ tgt,
                              const int* __restrict__ n1,
                              float* __restrict__ n12, int n) {
  const int i = blockIdx.x;
  const int t = threadIdx.x;
  int r = n1[i];
  r = (r < 0) ? 0 : ((r >= n) ? (n - 1) : r);
  const int tr = tgt[r];
  const float* __restrict__ row = dist + (size_t)r * n;

  float bmin = QL_BIG;
  for (int j = t; j < n; j += 256) {
    const float dv = row[j];
    if (tgt[j] != tr) bmin = fminf(bmin, dv);
  }
  __shared__ float s[256];
  s[t] = bmin;
  __syncthreads();
  for (int off = 128; off > 0; off >>= 1) {
    if (t < off) s[t] = fminf(s[t], s[t + off]);
    __syncthreads();
  }
  if (t == 0) n12[i] = s[0];
}

// ---------------------------------------------------------------------------
// Kernel 5: loss = mean(relu(ap-an+m)) + 0.1*mean(relu(ap-n12+m)); prec.
// ---------------------------------------------------------------------------
__global__ void ql_final_kernel(const float* __restrict__ ap,
                                const float* __restrict__ an,
                                const float* __restrict__ n12,
                                float* __restrict__ out, int n) {
  const int t = threadIdx.x;
  float s1 = 0.f, s2 = 0.f, pc = 0.f;
  for (int i = t; i < n; i += 256) {
    const float a = ap[i], b = an[i], c = n12[i];
    s1 += fmaxf(a - b + QL_MARGIN, 0.f);
    s2 += fmaxf(a - c + QL_MARGIN, 0.f);
    pc += (b > a) ? 1.f : 0.f;
  }
  __shared__ float x1[256], x2[256], x3[256];
  x1[t] = s1; x2[t] = s2; x3[t] = pc;
  __syncthreads();
  for (int off = 128; off > 0; off >>= 1) {
    if (t < off) { x1[t] += x1[t + off]; x2[t] += x2[t + off]; x3[t] += x3[t + off]; }
    __syncthreads();
  }
  if (t == 0) {
    const float inv = 1.0f / (float)n;
    out[0] = x1[0] * inv + 0.1f * (x2[0] * inv);  // loss
    out[1] = x3[0] * inv;                         // prec
  }
}

// ---------------------------------------------------------------------------
extern "C" void kernel_launch(void* const* d_in, const int* in_sizes, int n_in,
                              void* d_out, int out_size, void* d_ws, size_t ws_size,
                              hipStream_t stream) {
  const float* X   = (const float*)d_in[0];
  const int*   tgt = (const int*)d_in[1];
  const int n = in_sizes[1];          // 8192
  const int d = in_sizes[0] / n;      // 128

  float* out  = (float*)d_out;
  float* dist = out + 2;              // d_out = [loss, prec, dist(n*n)]

  // workspace layout
  char* ws = (char*)d_ws;
  bf16_t* Xhi = (bf16_t*)ws;  ws += (size_t)n * d * sizeof(bf16_t);
  bf16_t* Xlo = (bf16_t*)ws;  ws += (size_t)n * d * sizeof(bf16_t);
  float* sq  = (float*)ws;    ws += (size_t)n * sizeof(float);
  float* ap  = (float*)ws;    ws += (size_t)n * sizeof(float);
  float* an  = (float*)ws;    ws += (size_t)n * sizeof(float);
  float* n12 = (float*)ws;    ws += (size_t)n * sizeof(float);
  int*   n1  = (int*)ws;      ws += (size_t)n * sizeof(int);

  ql_prep_kernel<<<n, d, 0, stream>>>(X, Xhi, Xlo, sq, d);

  dim3 grid(n / 64, n / 32);
  ql_gemm_dist_kernel<<<grid, 256, 0, stream>>>(Xhi, Xlo, sq, dist, n, d);

  ql_row_reduce_kernel<<<n, 256, 0, stream>>>(dist, tgt, ap, an, n1, n);
  ql_n12_kernel<<<n, 256, 0, stream>>>(dist, tgt, n1, n12, n);
  ql_final_kernel<<<1, 256, 0, stream>>>(ap, an, n12, out, n);
}
